// StudentModelPLM_12197707120734
// MI455X (gfx1250) — matrix operand order repaired and checked by reading, not compiled, not run on hardware
//
#include <hip/hip_runtime.h>

// ---------------------------------------------------------------------------
// MI455X / gfx1250: wave32, v_wmma_f32_16x16x32_f16, async global->LDS staging.
// Shapes: B=32 N=256 H=768 HEADS=12 DH=64 PH=128 VOCAB=35
// ---------------------------------------------------------------------------

typedef _Float16 half_t;
typedef __attribute__((ext_vector_type(16))) _Float16 v16h;
typedef __attribute__((ext_vector_type(8)))  float    v8f;

#define B_     32
#define N_     256
#define H_     768
#define HEADS_ 12
#define DH_    64
#define PH_    128
#define VOC_   35

// LDS strides (halves), padded so the 16 column-lanes of a fragment spread
// across 16 LDS banks (dword-stride gcd with 64 == 4).
#define SAS 776   // A tile stride for K=768 panels
#define SBS 40    // staged B panel stride (32 K + 8 pad)
#define SKS 72    // k tile stride  (DH 64 + 8)
#define SVS 264   // v^T tile stride (N 256 + 8)
#define SQS 72    // q tile stride
#define SPS 264   // prob tile stride

// ---- CDNA5 async global->LDS copy (ASYNCcnt) ------------------------------
__device__ __forceinline__ void async_b128(unsigned lds_off, const void* gptr) {
  asm volatile("global_load_async_to_lds_b128 %0, %1, off"
               :: "v"(lds_off), "v"(gptr) : "memory");
}
__device__ __forceinline__ void wait_async0() {
  asm volatile("s_wait_asynccnt 0" ::: "memory");
}
__device__ __forceinline__ unsigned lds_off(const void* p) {
  return (unsigned)(unsigned long long)p;  // low 32 bits of flat addr = LDS offset
}

// ---- WMMA fragment helpers (ISA 7.12.2, wave32) ---------------------------
// A (16x32 f16): lane L<16 row M=L holds K {b..b+7, b+16..b+23}; lane L+16
// same row, +8. Both runs contiguous -> two b128 LDS loads.
__device__ __forceinline__ v16h load_a(const half_t* lds, int stride, int kbase, int lane) {
  const half_t* p = lds + (lane & 15) * stride + kbase + 8 * (lane >> 4);
  union { uint4 u[2]; v16h h; } c;
  c.u[0] = *reinterpret_cast<const uint4*>(p);
  c.u[1] = *reinterpret_cast<const uint4*>(p + 16);
  return c.h;
}
// B (32x16 f16) from K-contiguous [N][K] storage: lane L<16 column N=L holds
// K {0..7, 16..23}, lane L+16 holds K {8..15, 24..31} -> two b128 loads.
__device__ __forceinline__ v16h load_bT(const half_t* lds, int strideK, int nbase, int kbase, int lane) {
  const half_t* p = lds + (nbase + (lane & 15)) * strideK + kbase + 8 * (lane >> 4);
  union { uint4 u[2]; v16h h; } c;
  c.u[0] = *reinterpret_cast<const uint4*>(p);
  c.u[1] = *reinterpret_cast<const uint4*>(p + 16);
  return c.h;
}
// D (16x16 f32): lane l -> N=l&15, VGPR r -> M = r + 8*(l>>4).
__device__ __forceinline__ v8f wmma_f16(v16h a, v16h b, v8f c) {
  return __builtin_amdgcn_wmma_f32_16x16x32_f16(false, a, false, b, (short)0, c, false, false);
}

// ---------------------------------------------------------------------------
__global__ void k_cvt(const float* __restrict__ s, half_t* __restrict__ d, int n) {
  int i = blockIdx.x * blockDim.x + threadIdx.x;
  int stride = gridDim.x * blockDim.x;
  for (; i < n; i += stride) d[i] = (half_t)s[i];
}

// 768x768 f32 -> transposed f16 (W^T, [col][row])
__global__ __launch_bounds__(256) void k_cvt_t(const float* __restrict__ s, half_t* __restrict__ d) {
  __shared__ float tile[32][33];
  int bx = blockIdx.x % 24, by = blockIdx.x / 24;
  int r0 = by * 32, c0 = bx * 32;
  int tx = threadIdx.x & 31, ty = threadIdx.x >> 5;
  for (int yy = ty; yy < 32; yy += 8) tile[yy][tx] = s[(size_t)(r0 + yy) * H_ + c0 + tx];
  __syncthreads();
  for (int yy = ty; yy < 32; yy += 8) d[(size_t)(c0 + yy) * H_ + r0 + tx] = (half_t)tile[tx][yy];
}

// ---------------------------------------------------------------------------
// Vocab prep: LN(pos_table) (35x128) then SK/SV = LN @ Wsk/Wsv + b (35x64).
// Exact: LN is per-(i,j) vector and paths[i,j] = pos_table[pool[i,j]].
// ---------------------------------------------------------------------------
__global__ __launch_bounds__(256) void k_vocab(
    const float* __restrict__ pos, const float* __restrict__ g, const float* __restrict__ bta,
    const float* __restrict__ Wsk, const float* __restrict__ bsk,
    const float* __restrict__ Wsv, const float* __restrict__ bsv,
    float* __restrict__ SK, float* __restrict__ SV) {
  __shared__ float lnp[VOC_ * PH_];
  __shared__ float mu[VOC_], ri[VOC_];
  const int tid = threadIdx.x;
  if (tid < VOC_) {
    float s = 0.f;
    for (int p = 0; p < PH_; ++p) s += pos[tid * PH_ + p];
    float m = s * (1.f / PH_);
    float v = 0.f;
    for (int p = 0; p < PH_; ++p) { float dd = pos[tid * PH_ + p] - m; v += dd * dd; }
    mu[tid] = m;
    ri[tid] = rsqrtf(v * (1.f / PH_) + 1e-5f);
  }
  __syncthreads();
  for (int idx = tid; idx < VOC_ * PH_; idx += 256) {
    int t = idx >> 7, p = idx & 127;
    lnp[idx] = (pos[idx] - mu[t]) * ri[t] * g[p] + bta[p];
  }
  __syncthreads();
  for (int idx = tid; idx < VOC_ * DH_; idx += 256) {
    int t = idx >> 6, d = idx & 63;
    float a = bsk[d], b = bsv[d];
    for (int p = 0; p < PH_; ++p) {
      float x = lnp[t * PH_ + p];
      a += x * Wsk[p * DH_ + d];
      b += x * Wsv[p * DH_ + d];
    }
    SK[idx] = a;
    SV[idx] = b;
  }
}

// ---------------------------------------------------------------------------
// QKV projection: grid=(512 row-tiles, 3). 16 rows x 768 cols, K=768.
// Double-buffered async B panels from W^T; q scaled 0.125; v written
// TRANSPOSED ([bh][d][n]) so the attention out-GEMM gets K-contiguous B.
// ---------------------------------------------------------------------------
__global__ __launch_bounds__(256) void k_qkv(
    const half_t* __restrict__ A, const half_t* __restrict__ W3T,
    const float* __restrict__ bq, const float* __restrict__ bk, const float* __restrict__ bv,
    half_t* __restrict__ qo, half_t* __restrict__ ko, half_t* __restrict__ vo) {
  __shared__ __align__(16) half_t sA[16 * SAS];
  __shared__ __align__(16) half_t sB[2][H_ * SBS];
  const int tid = threadIdx.x, lane = tid & 31, wave = tid >> 5;
  const int rt = blockIdx.x, which = blockIdx.y;
  const half_t* Wm = W3T + (size_t)which * H_ * H_;  // [col][k]
  const float* bias = (which == 0) ? bq : (which == 1) ? bk : bv;
  half_t* out = (which == 0) ? qo : (which == 1) ? ko : vo;

  // A tile: 16 rows x 768, async, padded stride
  {
    const half_t* src = A + (size_t)rt * 16 * H_;
    for (int idx = tid; idx < 16 * 96; idx += 256) {
      int r = idx / 96, part = idx % 96;
      async_b128(lds_off(sA + r * SAS + part * 8), src + r * H_ + part * 8);
    }
  }
  // first B panel
  for (int idx = tid; idx < H_ * 4; idx += 256) {
    int n = idx >> 2, part = idx & 3;
    async_b128(lds_off(&sB[0][n * SBS + part * 8]), Wm + (size_t)n * H_ + part * 8);
  }
  wait_async0();
  __syncthreads();

  v8f acc[6] = {};
  const int n0 = wave * 96;
  for (int ks = 0; ks < 24; ++ks) {
    const int cur = ks & 1;
    if (ks + 1 < 24) {  // async-prefetch next panel into other buffer
      const half_t* src = Wm + (ks + 1) * 32;
      for (int idx = tid; idx < H_ * 4; idx += 256) {
        int n = idx >> 2, part = idx & 3;
        async_b128(lds_off(&sB[cur ^ 1][n * SBS + part * 8]), src + (size_t)n * H_ + part * 8);
      }
    }
    if (ks + 2 < 24) {  // L2 prefetch for the panel after that
      const char* nx = reinterpret_cast<const char*>(Wm + (ks + 2) * 32);
      for (int l = tid; l < 96; l += 256) __builtin_prefetch(nx + l * 128, 0, 0);
    }
    v16h a = load_a(sA, SAS, ks * 32, lane);
#pragma unroll
    for (int t = 0; t < 6; ++t) {
      v16h b = load_bT(sB[cur], SBS, n0 + t * 16, 0, lane);
      acc[t] = wmma_f16(a, b, acc[t]);
    }
    wait_async0();
    __syncthreads();
  }
#pragma unroll
  for (int t = 0; t < 6; ++t) {
#pragma unroll
    for (int r = 0; r < 8; ++r) {
      int m = r + 8 * (lane >> 4);
      int gr = rt * 16 + m;
      int bi = gr >> 8, n = gr & 255;
      int col = n0 + t * 16 + (lane & 15);
      int head = col >> 6, d = col & 63;
      float val = acc[t][r] + bias[col];
      if (which == 0) val *= 0.125f;
      size_t bh = (size_t)bi * HEADS_ + head;
      if (which == 2) out[(bh * DH_ + d) * N_ + n] = (half_t)val;      // v^T
      else            out[(bh * N_ + n) * DH_ + d] = (half_t)val;      // q,k
    }
  }
}

// ---------------------------------------------------------------------------
// Attention: one block per (b,h); k ([j][d]) and v^T ([d][j]) LDS-resident via
// async copies. Logits q@k^T + 35-entry struct gather, softmax, bucketed
// struct_v (ds_add_f32), P@V via WMMA. Writes probs (output 2) + attn f16.
// ---------------------------------------------------------------------------
__global__ __launch_bounds__(256) void k_att(
    const half_t* __restrict__ q, const half_t* __restrict__ k, const half_t* __restrict__ vT,
    const float* __restrict__ SK, const float* __restrict__ SV,
    const int* __restrict__ pool, const float* __restrict__ bias,
    float* __restrict__ w_out, half_t* __restrict__ attn_out) {
  __shared__ __align__(16) half_t sK[N_ * SKS];    // [j][d]: K-contig B for logits
  __shared__ __align__(16) half_t sVT[DH_ * SVS];  // [d][j]: K-contig B for out
  __shared__ float  sLog[16 * N_];
  __shared__ __align__(16) half_t sP[16 * SPS];
  __shared__ float  sSK[VOC_ * DH_];
  __shared__ float  sSV[VOC_ * DH_];
  __shared__ __align__(16) half_t sQ[16 * SQS];
  __shared__ float  sQSK[16 * 36];
  __shared__ float  sBkt[16 * 36];
  __shared__ float  sOutB[16 * DH_];
  __shared__ float  sOutS[16 * DH_];
  __shared__ float  sRed[16 * 16];
  __shared__ float  sMax[16], sSum[16];

  const int tid = threadIdx.x, lane = tid & 31, wave = tid >> 5;
  const int bh = blockIdx.x;
  const int b = bh / HEADS_, h = bh % HEADS_;
  const size_t base = (size_t)bh * N_ * DH_;

  for (int idx = tid; idx < N_ * 4; idx += 256) {     // k: 256 rows x 4 x b128
    int j = idx >> 2, part = idx & 3;
    async_b128(lds_off(sK + j * SKS + part * 8), k + base + j * DH_ + part * 8);
  }
  for (int idx = tid; idx < DH_ * 16; idx += 256) {   // v^T: 64 rows x 16 x b128
    int d = idx >> 4, part = idx & 15;
    async_b128(lds_off(sVT + d * SVS + part * 8), vT + base + d * N_ + part * 8);
  }
  for (int idx = tid; idx < VOC_ * DH_; idx += 256) { sSK[idx] = SK[idx]; sSV[idx] = SV[idx]; }
  wait_async0();
  __syncthreads();

  for (int it = 0; it < 16; ++it) {
    const int i0 = it * 16;
    for (int idx = tid; idx < 16 * 8; idx += 256) {   // q tile (b128, padded)
      int r = idx >> 3, part = idx & 7;
      async_b128(lds_off(sQ + r * SQS + part * 8), q + base + (i0 + r) * DH_ + part * 8);
    }
    wait_async0();
    __syncthreads();

    // logits: q(16x64) @ k^T(64x256); wave w -> n-tiles 2w, 2w+1
#pragma unroll
    for (int tt = 0; tt < 2; ++tt) {
      int nt = wave * 2 + tt;
      v8f acc = {};
#pragma unroll
      for (int ks = 0; ks < 2; ++ks) {
        v16h a = load_a(sQ, SQS, ks * 32, lane);
        v16h bf = load_bT(sK, SKS, nt * 16, ks * 32, lane);
        acc = wmma_f16(a, bf, acc);
      }
#pragma unroll
      for (int r = 0; r < 8; ++r) {
        int m = r + 8 * (lane >> 4);
        sLog[m * N_ + nt * 16 + (lane & 15)] = acc[r];
      }
    }
    // q . SK^T (16 x 35) + zero buckets
    for (int idx = tid; idx < 16 * 36; idx += 256) sBkt[idx] = 0.f;
    for (int task = tid; task < 16 * VOC_; task += 256) {
      int i = task / VOC_, t = task % VOC_;
      float s = 0.f;
      for (int d = 0; d < DH_; ++d) s += (float)sQ[i * SQS + d] * sSK[t * DH_ + d];
      sQSK[i * 36 + t] = s;
    }
    __syncthreads();

    // softmax: 16 threads per row
    {
      const int row = tid >> 4, c = tid & 15;
      const int gi = i0 + row;
      int tl[16];
#pragma unroll
      for (int u = 0; u < 16; ++u) tl[u] = pool[gi * N_ + c * 16 + u];
      float lmax = -3.0e38f;
#pragma unroll
      for (int u = 0; u < 16; ++u) {
        int j = c * 16 + u;
        float l = sLog[row * N_ + j] + sQSK[row * 36 + tl[u]] + bias[b * N_ + j];
        sLog[row * N_ + j] = l;
        lmax = fmaxf(lmax, l);
      }
      sRed[row * 16 + c] = lmax;
      __syncthreads();
      if (c == 0) {
        float m = -3.0e38f;
        for (int u = 0; u < 16; ++u) m = fmaxf(m, sRed[row * 16 + u]);
        sMax[row] = m;
      }
      __syncthreads();
      float m = sMax[row], s = 0.f;
#pragma unroll
      for (int u = 0; u < 16; ++u) {
        int j = c * 16 + u;
        float e = __expf(sLog[row * N_ + j] - m);
        sLog[row * N_ + j] = e;
        s += e;
      }
      sRed[row * 16 + c] = s;
      __syncthreads();
      if (c == 0) {
        float t2 = 0.f;
        for (int u = 0; u < 16; ++u) t2 += sRed[row * 16 + u];
        sSum[row] = t2;
      }
      __syncthreads();
      float inv = 1.f / sSum[row];
      size_t wbase = ((size_t)bh * N_ + gi) * N_;
#pragma unroll
      for (int u = 0; u < 16; ++u) {
        int j = c * 16 + u;
        float p = sLog[row * N_ + j] * inv;
        w_out[wbase + j] = p;
        sP[row * SPS + j] = (half_t)p;
        atomicAdd(&sBkt[row * 36 + tl[u]], p);
      }
    }
    __syncthreads();

    // out: waves 0-3 P(16x256)@V(256x64) WMMA; waves 4-7 buckets@SV (VALU)
    if (wave < 4) {
      int nt = wave;
      v8f acc = {};
#pragma unroll
      for (int ks = 0; ks < 8; ++ks) {
        v16h a = load_a(sP, SPS, ks * 32, lane);
        v16h bf = load_bT(sVT, SVS, nt * 16, ks * 32, lane);
        acc = wmma_f16(a, bf, acc);
      }
#pragma unroll
      for (int r = 0; r < 8; ++r) {
        int m = r + 8 * (lane >> 4);
        sOutB[m * DH_ + nt * 16 + (lane & 15)] = acc[r];
      }
    } else {
      int t2 = tid - 128;
      for (int u = 0; u < 8; ++u) {
        int idx = t2 * 8 + u;
        int i = idx >> 6, d = idx & 63;
        float s = 0.f;
        for (int t = 0; t < VOC_; ++t) s += sBkt[i * 36 + t] * sSV[t * DH_ + d];
        sOutS[idx] = s;
      }
    }
    __syncthreads();

    for (int idx = tid; idx < 16 * DH_; idx += 256) {
      int i = idx >> 6, d = idx & 63;
      size_t gr = (size_t)b * N_ + i0 + i;
      attn_out[gr * H_ + h * DH_ + d] = (half_t)(sOutB[idx] + sOutS[idx]);
    }
    __syncthreads();
  }
}

// ---------------------------------------------------------------------------
// Output projection (A @ Wo^T-staged) + ReLU + residual + LayerNorm fused.
// ---------------------------------------------------------------------------
__global__ __launch_bounds__(256) void k_out(
    const half_t* __restrict__ A, const half_t* __restrict__ WT,
    const float* __restrict__ bo, const float* __restrict__ nodes,
    const float* __restrict__ g, const float* __restrict__ bb,
    float* __restrict__ y) {
  __shared__ __align__(16) half_t sA[16 * SAS];
  __shared__ __align__(16) half_t sB[H_ * SBS];
  __shared__ float sC[16 * H_];
  __shared__ float sRed[16 * 16];
  __shared__ float sM[16], sR[16];

  const int tid = threadIdx.x, lane = tid & 31, wave = tid >> 5;
  const int rt = blockIdx.x;
  {
    const half_t* src = A + (size_t)rt * 16 * H_;
    for (int idx = tid; idx < 16 * 96; idx += 256) {
      int r = idx / 96, part = idx % 96;
      async_b128(lds_off(sA + r * SAS + part * 8), src + r * H_ + part * 8);
    }
  }
  v8f acc[6] = {};
  const int n0 = wave * 96;
  for (int ks = 0; ks < 24; ++ks) {
    for (int idx = tid; idx < H_ * 4; idx += 256) {
      int n = idx >> 2, part = idx & 3;
      async_b128(lds_off(sB + n * SBS + part * 8), WT + (size_t)n * H_ + ks * 32 + part * 8);
    }
    if (ks + 1 < 24) {
      const char* nx = reinterpret_cast<const char*>(WT + (ks + 1) * 32);
      for (int l = tid; l < 96; l += 256) __builtin_prefetch(nx + l * 128, 0, 0);
    }
    wait_async0();
    __syncthreads();
    v16h a = load_a(sA, SAS, ks * 32, lane);
#pragma unroll
    for (int t = 0; t < 6; ++t) {
      v16h bf = load_bT(sB, SBS, n0 + t * 16, 0, lane);
      acc[t] = wmma_f16(a, bf, acc[t]);
    }
    __syncthreads();
  }
#pragma unroll
  for (int t = 0; t < 6; ++t)
#pragma unroll
    for (int r = 0; r < 8; ++r) {
      int m = r + 8 * (lane >> 4);
      int col = n0 + t * 16 + (lane & 15);
      sC[m * H_ + col] = fmaxf(acc[t][r] + bo[col], 0.f);
    }
  __syncthreads();

  const int row = tid >> 4, c = tid & 15;
  const size_t gr = (size_t)rt * 16 + row;
  float s = 0.f;
  for (int u = 0; u < 48; ++u) {
    int col = c * 48 + u;
    float x = nodes[gr * H_ + col] + sC[row * H_ + col];
    sC[row * H_ + col] = x;
    s += x;
  }
  sRed[row * 16 + c] = s;
  __syncthreads();
  if (c == 0) {
    float t2 = 0.f;
    for (int u = 0; u < 16; ++u) t2 += sRed[row * 16 + u];
    sM[row] = t2 * (1.f / H_);
  }
  __syncthreads();
  float mean = sM[row];
  s = 0.f;
  for (int u = 0; u < 48; ++u) {
    int col = c * 48 + u;
    float d = sC[row * H_ + col] - mean;
    s += d * d;
  }
  sRed[row * 16 + c] = s;
  __syncthreads();
  if (c == 0) {
    float t2 = 0.f;
    for (int u = 0; u < 16; ++u) t2 += sRed[row * 16 + u];
    sR[row] = rsqrtf(t2 * (1.f / H_) + 1e-5f);
  }
  __syncthreads();
  float ri = sR[row];
  for (int u = 0; u < 48; ++u) {
    int col = c * 48 + u;
    y[gr * H_ + col] = (sC[row * H_ + col] - mean) * ri * g[col] + bb[col];
  }
}

// ---------------------------------------------------------------------------
extern "C" void kernel_launch(void* const* d_in, const int* in_sizes, int n_in,
                              void* d_out, int out_size, void* d_ws, size_t ws_size,
                              hipStream_t stream) {
  (void)in_sizes; (void)n_in; (void)out_size; (void)ws_size;
  const float* nodes = (const float*)d_in[0];
  const float* bias  = (const float*)d_in[1];
  const int*   pool  = (const int*)d_in[2];
  const float* pos   = (const float*)d_in[3];
  const float* Wq = (const float*)d_in[4];  const float* bq = (const float*)d_in[5];
  const float* Wk = (const float*)d_in[6];  const float* bk = (const float*)d_in[7];
  const float* Wv = (const float*)d_in[8];  const float* bv = (const float*)d_in[9];
  const float* Wsk = (const float*)d_in[10]; const float* bsk = (const float*)d_in[11];
  const float* Wsv = (const float*)d_in[12]; const float* bsv = (const float*)d_in[13];
  const float* Wo = (const float*)d_in[14]; const float* bo = (const float*)d_in[15];
  const float* g_path = (const float*)d_in[16]; const float* b_path = (const float*)d_in[17];
  const float* g_norm = (const float*)d_in[18]; const float* b_norm = (const float*)d_in[19];

  float* y_out = (float*)d_out;                          // (B,N,H)
  float* w_out = (float*)d_out + (size_t)B_ * N_ * H_;   // (B,12,N,N)

  char* ws = (char*)d_ws;
  size_t off = 0;
  auto alloc = [&](size_t bytes) {
    char* p = ws + off;
    off += (bytes + 255) & ~(size_t)255;
    return p;
  };
  const size_t NE = (size_t)B_ * N_ * H_;
  const size_t WE = (size_t)H_ * H_;
  half_t* nodesH = (half_t*)alloc(NE * 2);
  half_t* W3T    = (half_t*)alloc(3 * WE * 2);  // q,k,v transposed f16
  half_t* WoT    = (half_t*)alloc(WE * 2);      // o transposed f16
  half_t* qH     = (half_t*)alloc(NE * 2);
  half_t* kH     = (half_t*)alloc(NE * 2);
  half_t* vTH    = (half_t*)alloc(NE * 2);      // v transposed per (b,h)
  half_t* attnH  = (half_t*)alloc(NE * 2);
  float*  SK     = (float*)alloc(VOC_ * DH_ * 4);
  float*  SV     = (float*)alloc(VOC_ * DH_ * 4);

  k_cvt<<<2048, 256, 0, stream>>>(nodes, nodesH, (int)NE);
  k_cvt_t<<<576, 256, 0, stream>>>(Wq, W3T);
  k_cvt_t<<<576, 256, 0, stream>>>(Wk, W3T + WE);
  k_cvt_t<<<576, 256, 0, stream>>>(Wv, W3T + 2 * WE);
  k_cvt_t<<<576, 256, 0, stream>>>(Wo, WoT);
  k_vocab<<<1, 256, 0, stream>>>(pos, g_path, b_path, Wsk, bsk, Wsv, bsv, SK, SV);
  k_qkv<<<dim3(B_ * N_ / 16, 3), 256, 0, stream>>>(nodesH, W3T, bq, bk, bv, qH, kH, vTH);
  k_att<<<B_ * HEADS_, 256, 0, stream>>>(qH, kH, vTH, SK, SV, pool, bias, w_out, attnH);
  k_out<<<B_ * N_ / 16, 256, 0, stream>>>(attnH, WoT, bo, nodes, g_norm, b_norm, y_out);
}
